// MixtureNet_10771777978556
// MI455X (gfx1250) — compile-verified, matrix operand order checked by reading, not run on hardware
//
#include <hip/hip_runtime.h>
#include <math.h>

// ---------------------------------------------------------------------------
// MixtureNet forward for gfx1250 (MI455X).  B=32, N=256, D=128, NH=4, NI=8,
// NL=2, ING=31.  GEMM-heavy parts (D x D projections, pairwise 64->32 MLP)
// use v_wmma_f32_16x16x32_bf16; everything else is VALU/LDS.
// ---------------------------------------------------------------------------

typedef __attribute__((ext_vector_type(16))) __bf16 v16bf;
typedef __attribute__((ext_vector_type(8)))  float  v8f;

#define BB   32
#define NN   256
#define DD   128
#define NHH  4
#define NII  8
#define NLL  2
#define INGC 31
#define NPAIR 32640            // 256*255/2
#define PTILES 2040            // NPAIR/16

__device__ __forceinline__ float gelu_f(float x) {
    return 0.5f * x * (1.0f + erff(x * 0.70710678118654752440f));
}
__device__ __forceinline__ float sigmoid_f(float x) {
    return 1.0f / (1.0f + expf(-x));
}

// ---------------------------------------------------------------------------
// WMMA GEMM strip kernel:  Y[M,Nout] = act( X[M,K=128] @ W(^T) + bias )
//   WT=1 : W is [Nout,128] row-major (torch-style, X @ W.T)
//   WT=0 : W is [128,Nout] row-major (X @ W)
// One wave computes a 16 x (NT*16) strip: A fragment loaded once per k-step,
// reused across NT column tiles; K fully unrolled (4 steps) -> NT*4 WMMAs.
// A-fragment lane layout per CDNA5 ISA 7.12.2 (16-bit A 16x32):
//   lane l<16: row M=l, K = k0+{0..7} then k0+{16..23}
//   lane l+16: row M=l, K = k0+{8..15} then k0+{24..31}
// B mirrors A with M<->N; C/D v8f layout per bridge doc.
// grid = (Nout/(NT*16), M/16), block = 32.  act: 0=none, 1=gelu.
// ---------------------------------------------------------------------------
template<int NT, int WT>
__global__ __launch_bounds__(32) void gemm_strip(
    const float* __restrict__ X, const float* __restrict__ W,
    const float* __restrict__ bias, float* __restrict__ Y,
    int Nout, int act)
{
    constexpr int KK = 128;
    int col0 = blockIdx.x * (NT * 16);
    int tm   = blockIdx.y;
    int lane = threadIdx.x;
    int hlf = lane >> 4;        // 0 or 1
    int l   = lane & 15;
    int rowA = tm * 16 + l;

    const float* xrow = X + (size_t)rowA * KK;
    __builtin_prefetch(xrow, 0, 3);                       // global_prefetch_b8
    if (WT) __builtin_prefetch(W + (size_t)(col0 + l) * KK, 0, 3);

    v8f acc[NT];
    #pragma unroll
    for (int i = 0; i < NT; ++i) acc[i] = (v8f){};

    #pragma unroll
    for (int k0 = 0; k0 < KK; k0 += 32) {
        int base = k0 + hlf * 8;
        const float* xr = xrow + base;
        v16bf a;
        #pragma unroll
        for (int t = 0; t < 8; ++t) {
            a[t]     = (__bf16)xr[t];
            a[8 + t] = (__bf16)xr[16 + t];
        }
        #pragma unroll
        for (int nt = 0; nt < NT; ++nt) {
            v16bf bm;
            if (WT) {
                const float* wr = W + (size_t)(col0 + nt * 16 + l) * KK + base;
                #pragma unroll
                for (int t = 0; t < 8; ++t) {
                    bm[t]     = (__bf16)wr[t];
                    bm[8 + t] = (__bf16)wr[16 + t];
                }
            } else {
                const float* wr = W + (size_t)base * Nout + col0 + nt * 16 + l;
                #pragma unroll
                for (int t = 0; t < 8; ++t) {
                    bm[t]     = (__bf16)wr[(size_t)t * Nout];
                    bm[8 + t] = (__bf16)wr[(size_t)(16 + t) * Nout];
                }
            }
            acc[nt] = __builtin_amdgcn_wmma_f32_16x16x32_bf16(
                          false, a, false, bm, (short)0, acc[nt], false, false);
        }
    }
    #pragma unroll
    for (int nt = 0; nt < NT; ++nt) {
        int col = col0 + nt * 16 + l;
        float bv = bias ? bias[col] : 0.0f;
        #pragma unroll
        for (int r = 0; r < 8; ++r) {
            int row = tm * 16 + hlf * 8 + r;  // C/D: lanes 0-15 -> M=r, 16-31 -> M=8+r
            float v = acc[nt][r] + bv;
            if (act == 1) v = gelu_f(v);
            Y[(size_t)row * Nout + col] = v;
        }
    }
}

// ---------------------------------------------------------------------------
// Embedding + Weber-Fechner + mood/season.  grid (N, B), block DD.
// ---------------------------------------------------------------------------
__global__ void embed_kernel(
    const float* __restrict__ ing, const int* __restrict__ mood,
    const int* __restrict__ season,
    const float* __restrict__ W_in, const float* __restrict__ b_in,
    const float* __restrict__ wfW1, const float* __restrict__ wfb1,
    const float* __restrict__ wfW2, const float* __restrict__ wfb2,
    const float* __restrict__ Wc,  const float* __restrict__ bc,
    const float* __restrict__ Em,  const float* __restrict__ Es,
    float* __restrict__ h)
{
    int n = blockIdx.x, b = blockIdx.y, d = threadIdx.x;
    const float* row = ing + ((size_t)b * NN + n) * INGC;
    float s = b_in[d] + bc[d];
    #pragma unroll
    for (int i = 0; i < INGC; ++i) s += row[i] * W_in[i * DD + d];
    float xl = log1pf(fmaxf(row[22], 0.0f));
    float wf8[8];
    #pragma unroll
    for (int o = 0; o < 8; ++o) wf8[o] = wfb2[o];
    #pragma unroll
    for (int j = 0; j < 16; ++j) {
        float hj = gelu_f(xl * wfW1[j] + wfb1[j]);
        #pragma unroll
        for (int o = 0; o < 8; ++o) wf8[o] += hj * wfW2[j * 8 + o];
    }
    #pragma unroll
    for (int o = 0; o < 8; ++o) s += wf8[o] * Wc[o * DD + d];
    s += Em[mood[b] * DD + d] + Es[season[b] * DD + d];
    h[((size_t)b * NN + n) * DD + d] = s;
}

// ---------------------------------------------------------------------------
// LayerNorm over last dim (= blockDim.x, power of two); optional residual.
// ---------------------------------------------------------------------------
__global__ void ln_kernel(const float* __restrict__ x, const float* __restrict__ res,
                          const float* __restrict__ g, const float* __restrict__ be,
                          float* __restrict__ y)
{
    extern __shared__ float sh[];
    int row = blockIdx.x, t = threadIdx.x, len = blockDim.x;
    float v = x[(size_t)row * len + t];
    if (res) v += res[(size_t)row * len + t];
    sh[t] = v; __syncthreads();
    for (int s = len >> 1; s > 0; s >>= 1) { if (t < s) sh[t] += sh[t + s]; __syncthreads(); }
    float mean = sh[0] / len; __syncthreads();
    float d = v - mean;
    sh[t] = d * d; __syncthreads();
    for (int s = len >> 1; s > 0; s >>= 1) { if (t < s) sh[t] += sh[t + s]; __syncthreads(); }
    float var = sh[0] / len;
    y[(size_t)row * len + t] = d * rsqrtf(var + 1e-5f) * g[t] + be[t];
}

// ---------------------------------------------------------------------------
// Attention: one block per (qi, head, b). blockDim >= max(Lk,32).
// ---------------------------------------------------------------------------
__global__ void attn_kernel(const float* __restrict__ q, const float* __restrict__ k,
                            const float* __restrict__ v, float* __restrict__ out,
                            int Lq, int Lk)
{
    extern __shared__ float sm[];
    int qi = blockIdx.x, hd = blockIdx.y, b = blockIdx.z;
    int t = threadIdx.x;
    const float* qp = q + ((size_t)(b * Lq + qi)) * DD + hd * 32;
    if (t < Lk) {
        const float* kp = k + ((size_t)(b * Lk + t)) * DD + hd * 32;
        float d = 0.0f;
        #pragma unroll
        for (int c = 0; c < 32; ++c) d += qp[c] * kp[c];
        sm[t] = d * 0.17677669529663687f;   // 1/sqrt(32)
    }
    __syncthreads();
    if (t == 0) {
        float mx = sm[0];
        for (int i = 1; i < Lk; ++i) mx = fmaxf(mx, sm[i]);
        float s = 0.0f;
        for (int i = 0; i < Lk; ++i) { float e = expf(sm[i] - mx); sm[i] = e; s += e; }
        float inv = 1.0f / s;
        for (int i = 0; i < Lk; ++i) sm[i] *= inv;
    }
    __syncthreads();
    if (t < 32) {
        float acc = 0.0f;
        for (int i = 0; i < Lk; ++i)
            acc += sm[i] * v[((size_t)(b * Lk + i)) * DD + hd * 32 + t];
        out[((size_t)(b * Lq + qi)) * DD + hd * 32 + t] = acc;
    }
}

// ---------------------------------------------------------------------------
// triu_indices(256, k=1): ii/jj, row-major pair ordering.
// ---------------------------------------------------------------------------
__global__ void triu_init(int* __restrict__ ii, int* __restrict__ jj)
{
    int i = threadIdx.x;                      // 0..255
    if (i >= NN) return;
    int start = i * (2 * NN - i - 1) / 2;     // i*(511-i)/2
    for (int j = i + 1; j < NN; ++j) {
        int p = start + (j - i - 1);
        ii[p] = i; jj[p] = j;
    }
}

// ---------------------------------------------------------------------------
// Pre-pack sn_W1 [64,32] into per-lane WMMA B-fragments (bf16).
// frag index f = k0step*2 + ntile  (k0step in {0,1}, ntile in {0,1});
// layout: frag[(f*32 + lane)*16 + e]  -> one aligned v16bf per (f,lane).
// ---------------------------------------------------------------------------
__global__ void pack_w1(const float* __restrict__ W1, __bf16* __restrict__ frag)
{
    int idx = blockIdx.x * blockDim.x + threadIdx.x;   // 4*32
    if (idx >= 4 * 32) return;
    int f = idx / 32, lane = idx % 32;
    int k0 = (f >> 1) * 32, n0 = (f & 1) * 16;
    int hlf = lane >> 4, l = lane & 15;
    int base = k0 + hlf * 8;
    for (int t = 0; t < 8; ++t) {
        frag[idx * 16 + t]     = (__bf16)W1[(base + t)      * 32 + n0 + l];
        frag[idx * 16 + 8 + t] = (__bf16)W1[(base + 16 + t) * 32 + n0 + l];
    }
}

// ---------------------------------------------------------------------------
// Pairwise scorer: per (tile of 16 pairs, b) one wave.
// A = diff[16,64] built on the fly from pq[ii]-pk[jj]; B fragments come from
// the pre-packed table -> 4 WMMAs.  Epilogue: gelu, dot with sn_W2 via
// __shfl_xor reduction within each 16-lane half, tanh, deterministic tile sum.
// ---------------------------------------------------------------------------
__global__ __launch_bounds__(32) void pair_kernel(
    const float* __restrict__ pq, const float* __restrict__ pk,
    const int* __restrict__ ii, const int* __restrict__ jj,
    const __bf16* __restrict__ w1frag,
    const float* __restrict__ b1, const float* __restrict__ W2,
    const float* __restrict__ b2, float* __restrict__ tilesum)
{
    int tile = blockIdx.x, b = blockIdx.y;
    int lane = threadIdx.x;
    int hlf = lane >> 4, l = lane & 15;
    int p = tile * 16 + l;                         // this lane's pair row
    const float* qr = pq + ((size_t)b * NN + ii[p]) * 64;
    const float* kr = pk + ((size_t)b * NN + jj[p]) * 64;
    const v16bf* fr = (const v16bf*)w1frag;

    v8f c0 = {}, c1 = {};
    #pragma unroll
    for (int ks = 0; ks < 2; ++ks) {
        int base = ks * 32 + hlf * 8;
        v16bf a;
        #pragma unroll
        for (int t = 0; t < 8; ++t) {
            a[t]     = (__bf16)(qr[base + t]      - kr[base + t]);
            a[8 + t] = (__bf16)(qr[base + 16 + t] - kr[base + 16 + t]);
        }
        v16bf bm0 = fr[(ks * 2 + 0) * 32 + lane];
        v16bf bm1 = fr[(ks * 2 + 1) * 32 + lane];
        c0 = __builtin_amdgcn_wmma_f32_16x16x32_bf16(false, a, false, bm0, (short)0, c0, false, false);
        c1 = __builtin_amdgcn_wmma_f32_16x16x32_bf16(false, a, false, bm1, (short)0, c1, false, false);
    }
    float w2l = W2[l], w2h = W2[16 + l];
    float b1l = b1[l], b1h = b1[16 + l], b2v = b2[0];
    float part = 0.0f;
    #pragma unroll
    for (int r = 0; r < 8; ++r) {
        // row m = hlf*8 + r; columns n = l (c0) and 16+l (c1) live across the half
        float t = gelu_f(c0[r] + b1l) * w2l + gelu_f(c1[r] + b1h) * w2h;
        t += __shfl_xor(t, 1, 32);
        t += __shfl_xor(t, 2, 32);
        t += __shfl_xor(t, 4, 32);
        t += __shfl_xor(t, 8, 32);               // stays inside each 16-lane half
        part += tanhf(t + b2v);
    }
    part += __shfl_xor(part, 16, 32);            // combine halves (rows 0-7 + 8-15)
    if (lane == 0) tilesum[(size_t)b * PTILES + tile] = part;
}

__global__ void pair_reduce(const float* __restrict__ tilesum, float* __restrict__ out)
{
    __shared__ float sh[256];
    int b = blockIdx.x, t = threadIdx.x;
    float s = 0.0f;
    for (int i = t; i < PTILES; i += 256) s += tilesum[(size_t)b * PTILES + i];
    sh[t] = s; __syncthreads();
    for (int st = 128; st > 0; st >>= 1) { if (t < st) sh[t] += sh[t + st]; __syncthreads(); }
    if (t == 0) out[b] = sh[0] * (1.0f / (float)NPAIR);
}

// ---------------------------------------------------------------------------
// Small helpers: broadcast rows across batch, small dense layer, row softmax.
// ---------------------------------------------------------------------------
__global__ void bcast_rows(const float* __restrict__ src, float* __restrict__ dst,
                           int rows, int len)
{
    int idx = blockIdx.x * blockDim.x + threadIdx.x;
    int tot = BB * rows * len;
    if (idx >= tot) return;
    dst[idx] = src[idx % (rows * len)];
}

__global__ void dense_small(const float* __restrict__ X, const float* __restrict__ W,
                            const float* __restrict__ b, float* __restrict__ Y,
                            int M, int K, int N, int wt, int act)
{
    int idx = blockIdx.x * blockDim.x + threadIdx.x;
    if (idx >= M * N) return;
    int m = idx / N, n = idx % N;
    float acc = b ? b[n] : 0.0f;
    const float* xr = X + (size_t)m * K;
    for (int k = 0; k < K; ++k)
        acc += xr[k] * (wt ? W[(size_t)n * K + k] : W[(size_t)k * N + n]);
    if (act == 1) acc = gelu_f(acc);
    else if (act == 2) acc = sigmoid_f(acc);
    else if (act == 3) acc = tanhf(acc);
    Y[idx] = acc;
}

__global__ void softmax_rows(float* __restrict__ x, int Mrows, int n)
{
    int m = blockIdx.x * blockDim.x + threadIdx.x;
    if (m >= Mrows) return;
    float* r = x + (size_t)m * n;
    float mx = r[0];
    for (int i = 1; i < n; ++i) mx = fmaxf(mx, r[i]);
    float s = 0.0f;
    for (int i = 0; i < n; ++i) { r[i] = expf(r[i] - mx); s += r[i]; }
    float inv = 1.0f / s;
    for (int i = 0; i < n; ++i) r[i] *= inv;
}

// ---------------------------------------------------------------------------
// Host orchestration
// ---------------------------------------------------------------------------
static inline void launch_gemm(const float* X, const float* W, const float* b,
                               float* Y, int M, int N, int wt, int act,
                               hipStream_t s)
{
    // K is always 128 for these GEMMs
    if (wt) {
        gemm_strip<4, 1><<<dim3(N / 64, M / 16), 32, 0, s>>>(X, W, b, Y, N, act);
    } else if (N >= 64) {
        gemm_strip<4, 0><<<dim3(N / 64, M / 16), 32, 0, s>>>(X, W, b, Y, N, act);
    } else { // N == 32
        gemm_strip<2, 0><<<dim3(N / 32, M / 16), 32, 0, s>>>(X, W, b, Y, N, act);
    }
}

extern "C" void kernel_launch(void* const* d_in, const int* in_sizes, int n_in,
                              void* d_out, int out_size, void* d_ws, size_t ws_size,
                              hipStream_t stream)
{
    (void)in_sizes; (void)n_in; (void)out_size; (void)ws_size;
    const float* ing    = (const float*)d_in[0];
    const int*   mood   = (const int*)  d_in[1];
    const int*   season = (const int*)  d_in[2];
    // params, in setup_inputs dict order
    #define PP(i) ((const float*)d_in[3 + (i)])
    const float *W_in=PP(0), *b_in=PP(1), *wf_W1=PP(2), *wf_b1=PP(3), *wf_W2=PP(4),
                *wf_b2=PP(5), *Wc=PP(6), *bc=PP(7), *E_mood=PP(8), *E_season=PP(9),
                *isab_ind=PP(10), *qkv1_W=PP(11), *qkv1_b=PP(12), *out1_W=PP(13),
                *out1_b=PP(14), *qkv2_W=PP(15), *qkv2_b=PP(16), *out2_W=PP(17),
                *out2_b=PP(18), *ln1_g=PP(19), *ln1_b=PP(20), *ln2_g=PP(21),
                *ln2_b=PP(22), *pma_seed=PP(23), *pma_qkv_W=PP(24), *pma_qkv_b=PP(25),
                *pma_out_W=PP(26), *pma_out_b=PP(27), *pma_ln_g=PP(28), *pma_ln_b=PP(29),
                *pq_W=PP(30), *pq_b=PP(31), *pk_W=PP(32), *pk_b=PP(33),
                *sn_W1=PP(34), *sn_b1=PP(35), *sn_W2=PP(36), *sn_b2=PP(37),
                *hm_W1=PP(38), *hm_b1=PP(39), *hm_ln_g=PP(40), *hm_ln_b=PP(41),
                *hm_W2=PP(42), *hm_b2=PP(43), *ht_W1=PP(44), *ht_b1=PP(45),
                *ht_W2=PP(46), *ht_b2=PP(47), *hs_W1=PP(48), *hs_b1=PP(49),
                *hs_W2=PP(50), *hs_b2=PP(51), *ha_W1=PP(52), *ha_b1=PP(53),
                *ha_W2=PP(54), *ha_b2=PP(55), *hh_W1=PP(56), *hh_b1=PP(57),
                *hh_W2=PP(58), *hh_b2=PP(59);
    #undef PP

    float* out = (float*)d_out;
    const int BN = BB * NN;          // 8192
    const int BI = BB * NII;         // 256
    const int DSQ = DD * DD;         // 16384

    // workspace layout (floats); all block sizes are multiples of 8 floats,
    // so every sub-buffer keeps 32-byte alignment for v16bf loads.
    float* ws = (float*)d_ws;
    float* h      = ws;                       // BN*DD   = 1048576
    float* qb     = h      + (size_t)BN * DD;
    float* kb     = qb     + (size_t)BN * DD;
    float* vb     = kb     + (size_t)BN * DD;
    float* ab     = vb     + (size_t)BN * DD; // attention output
    float* pb     = ab     + (size_t)BN * DD; // out-proj output
    float* hhb    = pb     + (size_t)BN * DD; // BI*DD = 32768
    float* indB   = hhb    + (size_t)BI * DD; // 32768
    float* pqb    = indB   + (size_t)BI * DD; // BN*64 = 524288
    float* pkb    = pqb    + (size_t)BN * 64;
    float* pooled = pkb    + (size_t)BN * 64; // BB*DD = 4096
    float* seedB  = pooled + BB * DD;         // 4096
    float* t1     = seedB  + BB * DD;         // 4096 (head scratch)
    float* tsum   = t1     + BB * DD;         // BB*PTILES = 65280
    int*   iiA    = (int*)(tsum + (size_t)BB * PTILES);
    int*   jjA    = iiA + NPAIR;
    __bf16* w1frag = (__bf16*)(jjA + NPAIR);  // 4*32*16 bf16 = 4KB

    // ---- pair indices, packed W1 fragments, embedding -------------------
    triu_init<<<1, 256, 0, stream>>>(iiA, jjA);
    pack_w1<<<1, 128, 0, stream>>>(sn_W1, w1frag);
    embed_kernel<<<dim3(NN, BB), DD, 0, stream>>>(
        ing, mood, season, W_in, b_in, wf_W1, wf_b1, wf_W2, wf_b2,
        Wc, bc, E_mood, E_season, h);

    // ---- ISAB stack -----------------------------------------------------
    for (int l = 0; l < NLL; ++l) {
        const float* q1W = qkv1_W + (size_t)l * 3 * DSQ;
        const float* q1b = qkv1_b + (size_t)l * 3 * DD;
        const float* q2W = qkv2_W + (size_t)l * 3 * DSQ;
        const float* q2b = qkv2_b + (size_t)l * 3 * DD;

        bcast_rows<<<(BI * DD + 255) / 256, 256, 0, stream>>>(
            isab_ind + (size_t)l * NII * DD, indB, NII, DD);

        // MHA1: ind attends to h
        launch_gemm(indB, q1W,           q1b,           qb, BI, DD, 1, 0, stream);
        launch_gemm(h,    q1W + DSQ,     q1b + DD,      kb, BN, DD, 1, 0, stream);
        launch_gemm(h,    q1W + 2 * DSQ, q1b + 2 * DD,  vb, BN, DD, 1, 0, stream);
        attn_kernel<<<dim3(NII, NHH, BB), 256, 256 * sizeof(float), stream>>>(
            qb, kb, vb, ab, NII, NN);
        launch_gemm(ab, out1_W + (size_t)l * DSQ, out1_b + (size_t)l * DD,
                    pb, BI, DD, 1, 0, stream);
        ln_kernel<<<BI, DD, DD * sizeof(float), stream>>>(
            pb, indB, ln1_g + (size_t)l * DD, ln1_b + (size_t)l * DD, hhb);

        // MHA2: h attends to hh
        launch_gemm(h,   q2W,           q2b,          qb, BN, DD, 1, 0, stream);
        launch_gemm(hhb, q2W + DSQ,     q2b + DD,     kb, BI, DD, 1, 0, stream);
        launch_gemm(hhb, q2W + 2 * DSQ, q2b + 2 * DD, vb, BI, DD, 1, 0, stream);
        attn_kernel<<<dim3(NN, NHH, BB), 32, 32 * sizeof(float), stream>>>(
            qb, kb, vb, ab, NN, NII);
        launch_gemm(ab, out2_W + (size_t)l * DSQ, out2_b + (size_t)l * DD,
                    pb, BN, DD, 1, 0, stream);
        ln_kernel<<<BN, DD, DD * sizeof(float), stream>>>(
            pb, h, ln2_g + (size_t)l * DD, ln2_b + (size_t)l * DD, h);
    }

    // ---- pairwise interaction ------------------------------------------
    launch_gemm(h, pq_W, pq_b, pqb, BN, 64, 0, 0, stream);
    launch_gemm(h, pk_W, pk_b, pkb, BN, 64, 0, 0, stream);
    pair_kernel<<<dim3(PTILES, BB), 32, 0, stream>>>(
        pqb, pkb, iiA, jjA, w1frag, sn_b1, sn_W2, sn_b2, tsum);
    pair_reduce<<<BB, 256, 0, stream>>>(tsum, out + 3264);   // pair_scores [B]

    // ---- PMA pooling ----------------------------------------------------
    bcast_rows<<<(BB * DD + 255) / 256, 256, 0, stream>>>(pma_seed, seedB, 1, DD);
    launch_gemm(seedB, pma_qkv_W,           pma_qkv_b,          qb, BB, DD, 1, 0, stream);
    launch_gemm(h,     pma_qkv_W + DSQ,     pma_qkv_b + DD,     kb, BN, DD, 1, 0, stream);
    launch_gemm(h,     pma_qkv_W + 2 * DSQ, pma_qkv_b + 2 * DD, vb, BN, DD, 1, 0, stream);
    attn_kernel<<<dim3(1, NHH, BB), 256, 256 * sizeof(float), stream>>>(
        qb, kb, vb, ab, 1, NN);
    launch_gemm(ab, pma_out_W, pma_out_b, pb, BB, DD, 1, 0, stream);
    ln_kernel<<<BB, DD, DD * sizeof(float), stream>>>(
        pb, seedB, pma_ln_g, pma_ln_b, pooled);

    // ---- heads ----------------------------------------------------------
    // mixture -> out[0..704)
    launch_gemm(pooled, hm_W1, hm_b1, t1, BB, 64, 0, 1, stream);
    ln_kernel<<<BB, 64, 64 * sizeof(float), stream>>>(t1, nullptr, hm_ln_g, hm_ln_b, t1);
    dense_small<<<(BB * 22 + 255) / 256, 256, 0, stream>>>(
        t1, hm_W2, hm_b2, out + 0, BB, 64, 22, 0, 2);
    // temporal -> out[704..2816)
    launch_gemm(pooled, ht_W1, ht_b1, t1, BB, DD, 0, 1, stream);
    dense_small<<<(BB * 66 + 255) / 256, 256, 0, stream>>>(
        t1, ht_W2, ht_b2, out + 704, BB, DD, 66, 0, 2);
    // synergy -> out[2816..2848)
    launch_gemm(pooled, hs_W1, hs_b1, t1, BB, 32, 0, 1, stream);
    dense_small<<<1, 256, 0, stream>>>(t1, hs_W2, hs_b2, out + 2816, BB, 32, 1, 0, 2);
    // accord -> out[2848..3232)
    launch_gemm(pooled, ha_W1, ha_b1, t1, BB, 32, 0, 1, stream);
    dense_small<<<(BB * 12 + 255) / 256, 256, 0, stream>>>(
        t1, ha_W2, ha_b2, out + 2848, BB, 32, 12, 0, 0);
    softmax_rows<<<1, 32, 0, stream>>>(out + 2848, BB, 12);
    // hedonic -> out[3232..3264)
    launch_gemm(pooled, hh_W1, hh_b1, t1, BB, 32, 0, 1, stream);
    dense_small<<<1, 256, 0, stream>>>(t1, hh_W2, hh_b2, out + 3232, BB, 32, 1, 0, 3);
}
